// Transformer_44538810860205
// MI455X (gfx1250) — matrix operand order repaired and checked by reading, not compiled, not run on hardware
//
#include <hip/hip_runtime.h>
#include <hip/hip_bf16.h>

// ---------------------------------------------------------------------------
// MI455X / gfx1250 implementation of the 6-layer MICF transformer.
// Heavy math via v_wmma_f32_16x16x32_bf16 (wave32 WMMA); tile staging via
// global_load_async_to_lds_b128 (ASYNCcnt) where no transpose is needed.
// ---------------------------------------------------------------------------

typedef __bf16  v16bf __attribute__((ext_vector_type(16)));
typedef float   v8f   __attribute__((ext_vector_type(8)));

#define BB   32
#define NN   256
#define DIMM 1024
#define HH   16
#define DHH  64
#define MLPD 4096
#define MROWS (BB * NN)            // 8192 token rows
#define ATT_SCALE 0.125f           // DH^-0.5
#define GELU_C 0.7978845608028654f // sqrt(2/pi)

__device__ __forceinline__ v8f wmma_bf16(v16bf a, v16bf b, v8f c) {
  return __builtin_amdgcn_wmma_f32_16x16x32_bf16(
      /*neg_a=*/false, a, /*neg_b=*/false, b,
      /*c_mod=*/(short)0, c, /*reuse_a=*/false, /*reuse_b=*/false);
}

// Async copy of 32 contiguous bytes global -> LDS (per lane).
// LDS operand is the AS(3) byte offset = low 32 bits of the generic pointer.
// Tracked by ASYNCcnt; caller must s_wait_asynccnt before the barrier.
__device__ __forceinline__ void async_copy32(const __bf16* lds_dst,
                                             const __bf16* gsrc) {
  const unsigned l = (unsigned)(size_t)lds_dst;
  asm volatile("global_load_async_to_lds_b128 %0, %1, off"
               :: "v"(l), "v"(gsrc) : "memory");
  asm volatile("global_load_async_to_lds_b128 %0, %1, off offset:16"
               :: "v"(l), "v"(gsrc) : "memory");
}

__device__ __forceinline__ void async_wait0() {
  asm volatile("s_wait_asynccnt 0x0" ::: "memory");
}

// A-matrix 16x32 bf16 fragment (ISA 7.12.2): lane L: M=L&15, half=L>>4.
// Per lane: K = half*8 .. +7  and  K = 16+half*8 .. +7  -> two 16B chunks.
__device__ __forceinline__ v16bf a_frag(const __bf16* __restrict__ p, int ld) {
  const int lane = threadIdx.x & 31;
  const int m = lane & 15, hf = lane >> 4;
  const __bf16* r = p + (size_t)m * ld + hf * 8;
  union { float4 f[2]; v16bf v; } u;
  u.f[0] = *(const float4*)(r);
  u.f[1] = *(const float4*)(r + 16);
  return u.v;
}

// B-matrix 32x16 bf16 fragment read from an N-major (transposed) tile
// Bt[N][Kld]: lane L: N=L&15, K = half*8 .. +7 and K = 16+half*8 .. +7 (two 16B chunks).
__device__ __forceinline__ v16bf b_frag(const __bf16* __restrict__ bt,
                                        int n0, int k0, int ldk) {
  const int lane = threadIdx.x & 31;
  const int n = lane & 15, hf = lane >> 4;
  const __bf16* r = bt + (size_t)(n0 + n) * ldk + k0 + hf * 8;
  union { float4 f[2]; v16bf v; } u;
  u.f[0] = *(const float4*)(r);
  u.f[1] = *(const float4*)(r + 16);
  return u.v;
}

__device__ __forceinline__ float gelu_exact(float x) {
  return 0.5f * x * (1.0f + tanhf(GELU_C * (x + 0.044715f * x * x * x)));
}

// ---------------------------------------------------------------------------
// Generic batched WMMA GEMM: C[z] = A[z](MxK) * B[z](KxN)
//   + biasN[col] + biasM[row] -> optional GELU -> + resid -> Cf / Cbf.
// Block = 256 thr (8 waves); tile 128x128; per-wave 64x32 (4x2 v8f accums).
// A tile staged via async global->LDS; B tile transposed manually.
// ---------------------------------------------------------------------------
__global__ __launch_bounds__(256) void k_gemm(
    const __bf16* __restrict__ A, const __bf16* __restrict__ B,
    float* __restrict__ Cf, __bf16* __restrict__ Cbf,
    const float* __restrict__ biasN, const float* __restrict__ biasM,
    const float* __restrict__ resid, int gelu_flag,
    int M, int N, int K, int ldA, int ldB, int ldC,
    long long sA, long long sB, long long sC) {
  __shared__ __bf16 As[128 * 32];   // [m][k] row-major, 8KB
  __shared__ __bf16 Bt[128 * 32];   // [n][k] transposed, 8KB

  const int z = blockIdx.z;
  A += (size_t)z * sA;
  B += (size_t)z * sB;
  const size_t coff = (size_t)z * sC;

  const int tM = blockIdx.y * 128, tN = blockIdx.x * 128;
  const int wave = threadIdx.x >> 5, lane = threadIdx.x & 31;
  const int wm = (wave >> 2) * 64;  // wave M offset in tile
  const int wn = (wave & 3) * 32;   // wave N offset in tile

  v8f acc[4][2];
#pragma unroll
  for (int i = 0; i < 4; ++i)
#pragma unroll
    for (int j = 0; j < 2; ++j) acc[i][j] = (v8f){};

  const int arow = threadIdx.x >> 1, ahalf = (threadIdx.x & 1) * 16;
  const int bk = threadIdx.x >> 3,  bn = (threadIdx.x & 7) * 16;

  for (int k0 = 0; k0 < K; k0 += 32) {
    // stage A tile [128][32] via async global -> LDS (no VGPR round trip)
    const __bf16* ga = A + (size_t)(tM + arow) * ldA + k0 + ahalf;
    async_copy32(&As[arow * 32 + ahalf], ga);
    // stage B tile transposed [128 n][32 k] (manual: needs transpose)
    const __bf16* gb = B + (size_t)(k0 + bk) * ldB + tN + bn;
    union { float4 f; __bf16 h[8]; } u0, u1;
    u0.f = *(const float4*)(gb);
    u1.f = *(const float4*)(gb + 8);
#pragma unroll
    for (int i = 0; i < 8; ++i) {
      Bt[(bn + i) * 32 + bk]     = u0.h[i];
      Bt[(bn + 8 + i) * 32 + bk] = u1.h[i];
    }
    if (k0 + 32 < K) {                      // gfx1250 global_prefetch_b8
      __builtin_prefetch(ga + 32, 0, 1);
      __builtin_prefetch(gb + (size_t)32 * ldB, 0, 1);
    }
    async_wait0();                          // own async LDS writes landed
    __syncthreads();

    v16bf bf0 = b_frag(Bt, wn, 0, 32);
    v16bf bf1 = b_frag(Bt, wn + 16, 0, 32);
#pragma unroll
    for (int mf = 0; mf < 4; ++mf) {
      v16bf af = a_frag(&As[(wm + mf * 16) * 32], 32);
      acc[mf][0] = wmma_bf16(af, bf0, acc[mf][0]);
      acc[mf][1] = wmma_bf16(af, bf1, acc[mf][1]);
    }
    __syncthreads();
  }

  // epilogue: C frag element (r,lane): row = r + (lane>=16?8:0), col = lane&15
  const int rofs = (lane >= 16) ? 8 : 0;
  const int cofs = lane & 15;
#pragma unroll
  for (int mf = 0; mf < 4; ++mf)
#pragma unroll
    for (int nf = 0; nf < 2; ++nf)
#pragma unroll
      for (int r = 0; r < 8; ++r) {
        const int gr = tM + wm + mf * 16 + r + rofs;
        const int gc = tN + wn + nf * 16 + cofs;
        float v = acc[mf][nf][r];
        if (biasN) v += biasN[gc];
        if (biasM) v += biasM[gr];
        if (gelu_flag) v = gelu_exact(v);
        const size_t ci = coff + (size_t)gr * ldC + gc;
        if (resid) v += resid[ci];
        if (Cf)  Cf[ci] = v;
        if (Cbf) Cbf[ci] = (__bf16)v;
      }
}

// ---------------------------------------------------------------------------
// LayerNorm over DIM=1024, f32 in -> bf16 out. One block per row.
// ---------------------------------------------------------------------------
__global__ __launch_bounds__(256) void k_layernorm(
    const float* __restrict__ x, const float* __restrict__ g,
    const float* __restrict__ b, __bf16* __restrict__ out) {
  const int row = blockIdx.x;
  const float* xr = x + (size_t)row * DIMM;
  float4 v = ((const float4*)xr)[threadIdx.x];
  float s  = v.x + v.y + v.z + v.w;
  float s2 = v.x * v.x + v.y * v.y + v.z * v.z + v.w * v.w;
#pragma unroll
  for (int o = 16; o; o >>= 1) {
    s  += __shfl_xor(s,  o, 32);
    s2 += __shfl_xor(s2, o, 32);
  }
  __shared__ float w1[8], w2[8];
  if ((threadIdx.x & 31) == 0) { w1[threadIdx.x >> 5] = s; w2[threadIdx.x >> 5] = s2; }
  __syncthreads();
  s = 0.f; s2 = 0.f;
#pragma unroll
  for (int i = 0; i < 8; ++i) { s += w1[i]; s2 += w2[i]; }
  const float mean = s * (1.0f / DIMM);
  const float var  = s2 * (1.0f / DIMM) - mean * mean;
  const float rstd = rsqrtf(var + 1e-5f);
  float4 gg = ((const float4*)g)[threadIdx.x];
  float4 bb = ((const float4*)b)[threadIdx.x];
  __bf16* o = out + (size_t)row * DIMM + threadIdx.x * 4;
  o[0] = (__bf16)((v.x - mean) * rstd * gg.x + bb.x);
  o[1] = (__bf16)((v.y - mean) * rstd * gg.y + bb.y);
  o[2] = (__bf16)((v.z - mean) * rstd * gg.z + bb.z);
  o[3] = (__bf16)((v.w - mean) * rstd * gg.w + bb.w);
}

// ---------------------------------------------------------------------------
// Flash-style WMMA attention. Block per (batch, head): 8 waves x 32 q-rows.
// qkv: [8192][3072] bf16 (Q | K | V, head-major cols). out: [8192][1024] bf16.
// LDS: Kc 8KB + Vt 8KB + Pw 32KB = 48KB. K chunk staged async.
// ---------------------------------------------------------------------------
__global__ __launch_bounds__(256) void k_attention(
    const __bf16* __restrict__ qkv, __bf16* __restrict__ out) {
  __shared__ __bf16 Kc[64 * 64];        // [j][d]   (== Bt layout for Q*K^T)
  __shared__ __bf16 Vt[64 * 64];        // [d][j]   (Bt layout for P*V)
  __shared__ __bf16 Pw[8][32 * 64];     // per-wave probabilities

  const int bh = blockIdx.x;
  const int bat = bh >> 4, h = bh & 15;
  const int wave = threadIdx.x >> 5, lane = threadIdx.x & 31;
  const int q0 = wave * 32;
  const size_t row0 = (size_t)bat * NN;

  // Q fragments: 2 M-frags x 2 K-slabs, straight from global (bf16)
  v16bf aq[2][2];
  const __bf16* qbase = qkv + row0 * 3072 + h * DHH;
#pragma unroll
  for (int mf = 0; mf < 2; ++mf)
#pragma unroll
    for (int ks = 0; ks < 2; ++ks)
      aq[mf][ks] = a_frag(qbase + (size_t)(q0 + mf * 16) * 3072 + ks * 32, 3072);

  float mrun[2][8], lrun[2][8];
  v8f accO[2][4];
#pragma unroll
  for (int mf = 0; mf < 2; ++mf) {
#pragma unroll
    for (int r = 0; r < 8; ++r) { mrun[mf][r] = -1e30f; lrun[mf][r] = 0.f; }
#pragma unroll
    for (int nf = 0; nf < 4; ++nf) accO[mf][nf] = (v8f){};
  }

  const int rofs = (lane >= 16) ? 8 : 0;
  const int cofs = lane & 15;

  for (int c = 0; c < 4; ++c) {          // 4 key chunks of 64
    __syncthreads();
    {
      const int j = threadIdx.x >> 2;            // 0..63 key within chunk
      const int qd = (threadIdx.x & 3) * 16;     // 16 d-cols per thread
      const size_t krow = (row0 + c * 64 + j) * 3072 + h * DHH;
      // K chunk: straight row copy -> async global -> LDS
      async_copy32(&Kc[j * 64 + qd], qkv + krow + 1024 + qd);
      // V chunk: transposed -> manual scatter
      const __bf16* sv = qkv + krow + 2048 + qd;
      union { float4 f; __bf16 hh[8]; } u0, u1;
      u0.f = *(const float4*)(sv);
      u1.f = *(const float4*)(sv + 8);
#pragma unroll
      for (int i = 0; i < 8; ++i) {
        Vt[(qd + i) * 64 + j]     = u0.hh[i];
        Vt[(qd + 8 + i) * 64 + j] = u1.hh[i];
      }
    }
    async_wait0();
    __syncthreads();

    // S = Q * Kc^T  (32 x 64 per wave)
    v8f s[2][4];
#pragma unroll
    for (int mf = 0; mf < 2; ++mf)
#pragma unroll
      for (int nf = 0; nf < 4; ++nf) s[mf][nf] = (v8f){};
#pragma unroll
    for (int ks = 0; ks < 2; ++ks) {
      v16bf bk[4];
#pragma unroll
      for (int nf = 0; nf < 4; ++nf) bk[nf] = b_frag(Kc, nf * 16, ks * 32, 64);
#pragma unroll
      for (int mf = 0; mf < 2; ++mf)
#pragma unroll
        for (int nf = 0; nf < 4; ++nf)
          s[mf][nf] = wmma_bf16(aq[mf][ks], bk[nf], s[mf][nf]);
    }

    // online softmax (row reductions across 16-lane C-frag halves)
#pragma unroll
    for (int mf = 0; mf < 2; ++mf)
#pragma unroll
      for (int r = 0; r < 8; ++r) {
        float mx = -1e30f;
#pragma unroll
        for (int nf = 0; nf < 4; ++nf) {
          float t = s[mf][nf][r] * ATT_SCALE;
          s[mf][nf][r] = t;
          mx = fmaxf(mx, t);
        }
#pragma unroll
        for (int o = 8; o; o >>= 1) mx = fmaxf(mx, __shfl_xor(mx, o, 32));
        const float mn = fmaxf(mrun[mf][r], mx);
        const float f = __expf(mrun[mf][r] - mn);
        mrun[mf][r] = mn;
        float ps = 0.f;
#pragma unroll
        for (int nf = 0; nf < 4; ++nf) {
          float p = __expf(s[mf][nf][r] - mn);
          s[mf][nf][r] = p;
          ps += p;
        }
#pragma unroll
        for (int o = 8; o; o >>= 1) ps += __shfl_xor(ps, o, 32);
        lrun[mf][r] = lrun[mf][r] * f + ps;
#pragma unroll
        for (int nf = 0; nf < 4; ++nf) accO[mf][nf][r] *= f;
        const int m = mf * 16 + r + rofs;
#pragma unroll
        for (int nf = 0; nf < 4; ++nf)
          Pw[wave][m * 64 + nf * 16 + cofs] = (__bf16)s[mf][nf][r];
      }

    // O += P * V (same-wave LDS RAW kept in order by DScnt)
#pragma unroll
    for (int mf = 0; mf < 2; ++mf)
#pragma unroll
      for (int ks = 0; ks < 2; ++ks) {
        v16bf ap = a_frag(&Pw[wave][mf * 16 * 64 + ks * 32], 64);
#pragma unroll
        for (int nf = 0; nf < 4; ++nf)
          accO[mf][nf] = wmma_bf16(ap, b_frag(Vt, nf * 16, ks * 32, 64),
                                   accO[mf][nf]);
      }
  }

  // normalize + store bf16 [token][h*64+d]
#pragma unroll
  for (int mf = 0; mf < 2; ++mf)
#pragma unroll
    for (int nf = 0; nf < 4; ++nf)
#pragma unroll
      for (int r = 0; r < 8; ++r) {
        const float o = accO[mf][nf][r] / lrun[mf][r];
        const int m = q0 + mf * 16 + r + rofs;
        const int d = nf * 16 + cofs;
        out[(row0 + m) * DIMM + h * DHH + d] = (__bf16)o;
      }
}

// ---------------------------------------------------------------------------
// Helper conversion / packing kernels
// ---------------------------------------------------------------------------
__global__ void k_cvt(const float* __restrict__ s, __bf16* __restrict__ d,
                      size_t n) {
  size_t i = (size_t)blockIdx.x * blockDim.x + threadIdx.x;
  if (i < n) d[i] = (__bf16)s[i];
}

__global__ void k_cvt_batched(const float* __restrict__ s,
                              __bf16* __restrict__ d, size_t cnt,
                              size_t ss, size_t ds) {
  size_t i = (size_t)blockIdx.x * blockDim.x + threadIdx.x;
  const size_t z = blockIdx.y;
  if (i < cnt) d[z * ds + i] = (__bf16)s[z * ss + i];
}

// pack conv-cat weight w[o][c][2] (f32) -> bf16 [o][2C] = [w0 | w1]
__global__ void k_pack_w(const float* __restrict__ w, __bf16* __restrict__ o,
                         int OC, int C) {
  int i = blockIdx.x * 256 + threadIdx.x;
  if (i < OC * C) {
    int oo = i / C, c = i % C;
    o[(size_t)oo * 2 * C + c]     = (__bf16)w[((size_t)oo * C + c) * 2];
    o[(size_t)oo * 2 * C + C + c] = (__bf16)w[((size_t)oo * C + c) * 2 + 1];
  }
}

// ---------------------------------------------------------------------------
// Host orchestration
// ---------------------------------------------------------------------------
static void launch_gemm(hipStream_t st, const __bf16* A, const __bf16* B,
                        float* Cf, __bf16* Cbf, const float* bN,
                        const float* bM, const float* resid, int gelu,
                        int M, int N, int K, int ldA, int ldB, int ldC,
                        long long sA = 0, long long sB = 0, long long sC = 0,
                        int batch = 1) {
  dim3 g((unsigned)(N / 128), (unsigned)(M / 128), (unsigned)batch);
  k_gemm<<<g, 256, 0, st>>>(A, B, Cf, Cbf, bN, bM, resid, gelu, M, N, K, ldA,
                            ldB, ldC, sA, sB, sC);
}

static void launch_cvt(hipStream_t st, const float* s, __bf16* d, size_t n) {
  k_cvt<<<(unsigned)((n + 255) / 256), 256, 0, st>>>(s, d, n);
}

extern "C" void kernel_launch(void* const* d_in, const int* in_sizes, int n_in,
                              void* d_out, int out_size, void* d_ws,
                              size_t ws_size, hipStream_t stream) {
  (void)in_sizes; (void)n_in; (void)out_size; (void)ws_size;
  const float* x_in   = (const float*)d_in[0];
  const float* ln1_g  = (const float*)d_in[1];
  const float* ln1_b  = (const float*)d_in[2];
  const float* Wqkv   = (const float*)d_in[3];
  const float* Wout   = (const float*)d_in[4];
  const float* bout   = (const float*)d_in[5];
  const float* ln2_g  = (const float*)d_in[6];
  const float* ln2_b  = (const float*)d_in[7];
  const float* W1     = (const float*)d_in[8];
  const float* b1     = (const float*)d_in[9];
  const float* W2     = (const float*)d_in[10];
  const float* b2     = (const float*)d_in[11];
  const float* conv1w = (const float*)d_in[12];
  const float* conv1b = (const float*)d_in[13];
  const float* skipw  = (const float*)d_in[14];
  const float* skipb  = (const float*)d_in[15];
  const float* fuw    = (const float*)d_in[16];
  const float* fub    = (const float*)d_in[17];

  const size_t S_ACT = (size_t)MROWS * DIMM * 4;   // 32 MB f32
  char* wp = (char*)d_ws;
  auto alloc = [&](size_t bytes) -> char* {
    char* p = wp;
    wp += (bytes + 255) & ~(size_t)255;
    return p;
  };
  float*  actA  = (float*)alloc(S_ACT);
  float*  actB  = (float*)alloc(S_ACT);
  float*  hist0 = (float*)alloc(S_ACT);
  float*  hist1 = (float*)alloc(S_ACT);
  __bf16* lnbuf = (__bf16*)alloc((size_t)MROWS * DIMM * 2);
  __bf16* qkvb  = (__bf16*)alloc((size_t)MROWS * 3072 * 2);
  __bf16* attnb = (__bf16*)alloc((size_t)MROWS * DIMM * 2);
  __bf16* mlpb  = (__bf16*)alloc((size_t)MROWS * MLPD * 2);
  __bf16* wqb   = (__bf16*)alloc((size_t)DIMM * 3072 * 2);
  __bf16* wob   = (__bf16*)alloc((size_t)DIMM * DIMM * 2);
  __bf16* w1b   = (__bf16*)alloc((size_t)DIMM * MLPD * 2);
  __bf16* w2b   = (__bf16*)alloc((size_t)MLPD * DIMM * 2);
  __bf16* xcat1 = (__bf16*)alloc((size_t)BB * 512 * DIMM * 2);
  __bf16* xcat2 = (__bf16*)alloc((size_t)BB * 512 * DIMM * 2);
  __bf16* x1a   = (__bf16*)alloc((size_t)MROWS * DIMM * 2);
  __bf16* fuWb  = (__bf16*)alloc((size_t)NN * 512 * 2);
  __bf16* skWb  = (__bf16*)alloc((size_t)NN * 512 * 2);
  __bf16* c1Wb  = (__bf16*)alloc((size_t)NN * NN * 2);

  float* hist[2] = {hist0, hist1};
  float* x = actA;
  float* spare = actB;

  hipMemcpyAsync(x, x_in, S_ACT, hipMemcpyDeviceToDevice, stream);
  launch_cvt(stream, conv1w, c1Wb, (size_t)NN * NN);  // shared across layers

  const size_t CNT = (size_t)NN * DIMM;   // per-batch activation slab (262144)
  const dim3 cg((unsigned)((CNT + 255) / 256), BB, 1);

  for (int nl = 0; nl < 6; ++nl) {
    if (nl > 1) {
      // ---- mixing: x1 = conv1(fu_cat(last1,last2)); x = skip_cat(x, x1)
      k_pack_w<<<(NN * NN + 255) / 256, 256, 0, stream>>>(
          fuw + (size_t)(nl - 2) * NN * NN * 2, fuWb, NN, NN);
      k_pack_w<<<(NN * NN + 255) / 256, 256, 0, stream>>>(
          skipw + (size_t)(nl - 2) * NN * NN * 2, skWb, NN, NN);
      k_cvt_batched<<<cg, 256, 0, stream>>>(hist[(nl - 1) & 1], xcat1, CNT,
                                            CNT, (size_t)512 * DIMM);
      k_cvt_batched<<<cg, 256, 0, stream>>>(hist[nl & 1],
                                            xcat1 + (size_t)NN * DIMM, CNT,
                                            CNT, (size_t)512 * DIMM);
      launch_gemm(stream, fuWb, xcat1, nullptr, x1a, nullptr,
                  fub + (size_t)(nl - 2) * NN, nullptr, 0,
                  NN, DIMM, 512, 512, DIMM, DIMM,
                  0, (long long)512 * DIMM, (long long)NN * DIMM, BB);
      k_cvt_batched<<<cg, 256, 0, stream>>>(x, xcat2, CNT, CNT,
                                            (size_t)512 * DIMM);
      launch_gemm(stream, c1Wb, x1a, nullptr, xcat2 + (size_t)NN * DIMM,
                  nullptr, conv1b, nullptr, 0,
                  NN, DIMM, NN, NN, DIMM, DIMM,
                  0, (long long)NN * DIMM, (long long)512 * DIMM, BB);
      hipMemcpyAsync(hist[nl & 1], x, S_ACT, hipMemcpyDeviceToDevice, stream);
      launch_gemm(stream, skWb, xcat2, spare, nullptr, nullptr,
                  skipb + (size_t)(nl - 2) * NN, nullptr, 0,
                  NN, DIMM, 512, 512, DIMM, DIMM,
                  0, (long long)512 * DIMM, (long long)NN * DIMM, BB);
      float* t = x; x = spare; spare = t;
    } else {
      hipMemcpyAsync(hist[nl & 1], x, S_ACT, hipMemcpyDeviceToDevice, stream);
    }

    // ---- attention block
    k_layernorm<<<MROWS, 256, 0, stream>>>(x, ln1_g + (size_t)nl * DIMM,
                                           ln1_b + (size_t)nl * DIMM, lnbuf);
    launch_cvt(stream, Wqkv + (size_t)nl * DIMM * 3072, wqb,
               (size_t)DIMM * 3072);
    launch_gemm(stream, lnbuf, wqb, nullptr, qkvb, nullptr, nullptr, nullptr,
                0, MROWS, 3072, DIMM, DIMM, 3072, 3072);
    k_attention<<<BB * HH, 256, 0, stream>>>(qkvb, attnb);
    launch_cvt(stream, Wout + (size_t)nl * DIMM * DIMM, wob,
               (size_t)DIMM * DIMM);
    launch_gemm(stream, attnb, wob, spare, nullptr,
                bout + (size_t)nl * DIMM, nullptr, x, 0,
                MROWS, DIMM, DIMM, DIMM, DIMM, DIMM);
    { float* t = x; x = spare; spare = t; }

    // ---- feed-forward block
    k_layernorm<<<MROWS, 256, 0, stream>>>(x, ln2_g + (size_t)nl * DIMM,
                                           ln2_b + (size_t)nl * DIMM, lnbuf);
    launch_cvt(stream, W1 + (size_t)nl * DIMM * MLPD, w1b,
               (size_t)DIMM * MLPD);
    launch_gemm(stream, lnbuf, w1b, nullptr, mlpb,
                b1 + (size_t)nl * MLPD, nullptr, nullptr, 1,
                MROWS, MLPD, DIMM, DIMM, MLPD, MLPD);
    launch_cvt(stream, W2 + (size_t)nl * MLPD * DIMM, w2b,
               (size_t)MLPD * DIMM);
    launch_gemm(stream, mlpb, w2b, spare, nullptr,
                b2 + (size_t)nl * DIMM, nullptr, x, 0,
                MROWS, DIMM, MLPD, MLPD, DIMM, DIMM);
    { float* t = x; x = spare; spare = t; }
  }

  hipMemcpyAsync(d_out, x, S_ACT, hipMemcpyDeviceToDevice, stream);
}
